// Cross_WindowAttention_309237645722
// MI455X (gfx1250) — matrix operand order repaired and checked by reading, not compiled, not run on hardware
//
#include <hip/hip_runtime.h>
#include <hip/hip_bf16.h>

typedef __attribute__((ext_vector_type(16))) __bf16 v16bf;
typedef __attribute__((ext_vector_type(8)))  float  v8f;

struct Frag32 { uint4 lo, hi; };

__device__ __forceinline__ unsigned short f2bf(float f) {
  unsigned int u = __float_as_uint(f);
  u = (u + 0x7FFFu + ((u >> 16) & 1u)) >> 16;   // round-to-nearest-even
  return (unsigned short)u;
}
__device__ __forceinline__ unsigned int pack2bf(float a, float b) {
  return (unsigned int)f2bf(a) | ((unsigned int)f2bf(b) << 16);
}

// A-operand fragment (16x32 bf16). LDS holds A rows contiguously (strideU32 dwords/row).
// Lanes 0-15 -> K 0..15, lanes 16-31 -> K 16..31 (contiguous, matches frag_b).
__device__ __forceinline__ v16bf frag_a(const unsigned int* lds, int row0, int strideU32, int kd0) {
  const int lane = threadIdx.x & 31, hf = lane >> 4;
  const unsigned int* p = lds + (row0 + (lane & 15)) * strideU32 + kd0 + hf * 8;
  Frag32 f;
  f.lo = *(const uint4*)(p);
  f.hi = *(const uint4*)(p + 4);
  return __builtin_bit_cast(v16bf, f);
}
// B-operand fragment (32x16 bf16); LDS holds B^T rows (one column of B contiguous over K).
// Column = lane&15; lanes 0-15 hold K 0..15, lanes 16-31 hold K 16..31.
__device__ __forceinline__ v16bf frag_b(const unsigned int* lds, int row0, int strideU32, int kd0) {
  const int lane = threadIdx.x & 31, hf = lane >> 4;
  const unsigned int* p = lds + (row0 + (lane & 15)) * strideU32 + kd0 + hf * 8;
  Frag32 f;
  f.lo = *(const uint4*)(p);
  f.hi = *(const uint4*)(p + 4);
  return __builtin_bit_cast(v16bf, f);
}
// B-operand fragment gathered from row-major [K x Ncols] LDS (used for V).
__device__ __forceinline__ v16bf frag_b_strided(const unsigned short* lds, int n, int strideU16, int k0) {
  const int lane = threadIdx.x & 31, hf = lane >> 4;
  const __bf16* p = (const __bf16*)lds;
  v16bf f;
#pragma unroll
  for (int e = 0; e < 16; ++e)
    f[e] = p[(k0 + hf * 16 + e) * strideU16 + n];
  return f;
}

__device__ __forceinline__ v8f wmma_bf16(v16bf a, v16bf b, v8f c) {
  return __builtin_amdgcn_wmma_f32_16x16x32_bf16(false, a, false, b, (short)0, c, false, false);
}

// Shared 128x128 MMA step over staged LDS tiles (8 waves: 4(M) x 2(N)).
__device__ __forceinline__ void mma_128x128(const unsigned int* lds_a, const unsigned int* lds_b,
                                            int wm, int wn, v8f acc[2][4]) {
  v16bf af[2], bfr[4];
#pragma unroll
  for (int mt = 0; mt < 2; ++mt) af[mt] = frag_a(lds_a, wm * 32 + mt * 16, 20, 0);
#pragma unroll
  for (int nt = 0; nt < 4; ++nt) bfr[nt] = frag_b(lds_b, wn * 64 + nt * 16, 20, 0);
#pragma unroll
  for (int mt = 0; mt < 2; ++mt)
#pragma unroll
    for (int nt = 0; nt < 4; ++nt)
      acc[mt][nt] = wmma_bf16(af[mt], bfr[nt], acc[mt][nt]);
}

// --- tile loaders (all-32-bit element offsets; every tensor < 2 GiB, so the
// backend can use SGPR-base + 32-bit VGPR-offset global addressing) ---------
__device__ __forceinline__ void load_a_fast(const float* __restrict__ A, unsigned lda, unsigned M0,
                                            unsigned Kc, unsigned int* lds_a, int t) {
#pragma unroll
  for (int p = 0; p < 4; ++p) {
    int lin = p * 256 + t;
    int m = lin >> 3, q4 = lin & 7;
    unsigned off = (M0 + m) * lda + Kc + q4 * 4;
    const float4 v = *(const float4*)(A + off);
    lds_a[m * 20 + q4 * 2]     = pack2bf(v.x, v.y);
    lds_a[m * 20 + q4 * 2 + 1] = pack2bf(v.z, v.w);
  }
}
__device__ __forceinline__ void load_a_guard(const float* __restrict__ A, unsigned lda, unsigned K,
                                             unsigned M0, unsigned Kc, unsigned int* lds_a, int t) {
#pragma unroll
  for (int p = 0; p < 8; ++p) {
    int lin = p * 256 + t;
    int m = lin >> 4, kd = lin & 15;
    unsigned k = Kc + kd * 2;
    unsigned off = (M0 + m) * lda + k;
    float f0 = (k < K) ? A[off] : 0.f;
    float f1 = (k + 1 < K) ? A[off + 1] : 0.f;
    lds_a[m * 20 + kd] = pack2bf(f0, f1);
  }
}
__device__ __forceinline__ void load_b_fast(const float* __restrict__ W, unsigned ldw, unsigned N0,
                                            unsigned Kc, unsigned int* lds_b, int t) {
  unsigned short* sb = (unsigned short*)lds_b;
#pragma unroll
  for (int p = 0; p < 8; ++p) {
    int lin = p * 256 + t;
    int k = lin >> 6, n2 = lin & 63;
    unsigned off = (Kc + k) * ldw + N0 + n2 * 2;
    const float2 v = *(const float2*)(W + off);
    sb[(n2 * 2) * 40 + k]     = f2bf(v.x);
    sb[(n2 * 2 + 1) * 40 + k] = f2bf(v.y);
  }
}
__device__ __forceinline__ void load_b_guard(const float* __restrict__ W, unsigned ldw, unsigned K,
                                             unsigned N0, unsigned Kc, unsigned int* lds_b, int t) {
  unsigned short* sb = (unsigned short*)lds_b;
#pragma unroll
  for (int p = 0; p < 8; ++p) {
    int lin = p * 256 + t;
    int k = lin >> 6, n2 = lin & 63;
    float2 v = make_float2(0.f, 0.f);
    if (Kc + k < K) v = *(const float2*)(W + (Kc + k) * ldw + N0 + n2 * 2);
    sb[(n2 * 2) * 40 + k]     = f2bf(v.x);
    sb[(n2 * 2 + 1) * 40 + k] = f2bf(v.y);
  }
}

// ---------------------------------------------------------------------------
// Kernel 1: gather relative-position bias -> rpb[H][64][64] (f32)
// ---------------------------------------------------------------------------
__global__ void rpb_gather(const float* __restrict__ table, const int* __restrict__ ridx,
                           float* __restrict__ rpb) {
  int ij = blockIdx.x * 256 + threadIdx.x;
  if (ij < 4096) {
    int r = ridx[ij];
#pragma unroll
    for (int h = 0; h < 16; ++h) rpb[h * 4096 + ij] = table[r * 16 + h];
  }
}

// ---------------------------------------------------------------------------
// Kernel 2: fused qkv = x@w1+b1 (+ y@w2+b2 for k/v tiles), scatter to
// q/k/v [B][H][N][hd] bf16. M=131072, N=1536. Block tile 128x128, 8 waves.
// ---------------------------------------------------------------------------
__global__ __launch_bounds__(256) void qkv_gemm(
    const float* __restrict__ x, const float* __restrict__ y,
    const float* __restrict__ w1, const float* __restrict__ b1,
    const float* __restrict__ w2, const float* __restrict__ b2,
    unsigned short* __restrict__ qws, unsigned short* __restrict__ kws,
    unsigned short* __restrict__ vws) {
  const unsigned N0 = (blockIdx.x % 12) * 128;
  const unsigned M0 = (blockIdx.x / 12) * 128;
  const bool inc_y = (N0 >= 512);      // q columns use only the x-path

  __shared__ __align__(16) unsigned int lds_a[128 * 20];  // 128 rows x 32 bf16 (stride 80B)
  __shared__ __align__(16) unsigned int lds_b[128 * 20];  // 128 cols x 32 bf16 (B^T rows)

  const int t = threadIdx.x;
  const int wave = t >> 5, wm = wave >> 1, wn = wave & 1;
  const int lane = t & 31, hf = lane >> 4, l16 = lane & 15;

  v8f acc[2][4];
#pragma unroll
  for (int mt = 0; mt < 2; ++mt)
#pragma unroll
    for (int nt = 0; nt < 4; ++nt) acc[mt][nt] = (v8f){};

#pragma unroll 1
  for (int ph = 0; ph < 2; ++ph) {
    if (ph == 1 && !inc_y) break;
    const float* A = ph ? y : x;
    const float* W = ph ? w2 : w1;
    const unsigned K = ph ? 1000u : 512u;
    const unsigned nfull = K / 32;     // 16 (x) or 31 (y): unguarded chunks
#pragma unroll 1
    for (unsigned kc = 0; kc < nfull; ++kc) {
      __syncthreads();
      load_a_fast(A, K, M0, kc * 32, lds_a, t);
      load_b_fast(W, 1536, N0, kc * 32, lds_b, t);
      __syncthreads();
      mma_128x128(lds_a, lds_b, wm, wn, acc);
    }
    if (K % 32) {                      // single guarded remainder chunk (y only)
      __syncthreads();
      load_a_guard(A, K, K, M0, nfull * 32, lds_a, t);
      load_b_guard(W, 1536, K, N0, nfull * 32, lds_b, t);
      __syncthreads();
      mma_128x128(lds_a, lds_b, wm, wn, acc);
    }
  }

  const float scale = 0.17677669529663687f;  // hd^-0.5
#pragma unroll
  for (int mt = 0; mt < 2; ++mt)
#pragma unroll
    for (int nt = 0; nt < 4; ++nt)
#pragma unroll
      for (int e = 0; e < 8; ++e) {
        unsigned m = M0 + wm * 32 + mt * 16 + e + 8 * hf;
        unsigned c = N0 + wn * 64 + nt * 16 + l16;
        float val = acc[mt][nt][e] + b1[c] + (inc_y ? b2[c] : 0.f);
        unsigned three = c >> 9, h = (c >> 5) & 15, d = c & 31;
        unsigned b = m >> 6, np = m & 63;
        unsigned idx = (((b * 16 + h) * 64) + np) * 32 + d;
        unsigned short* dst = (three == 0) ? qws : (three == 1) ? kws : vws;
        if (three == 0) val *= scale;
        dst[idx] = f2bf(val);
      }
}

// ---------------------------------------------------------------------------
// Kernel 3: attention per (b,h). 4 waves, wave w handles rows 16w..16w+15.
// ---------------------------------------------------------------------------
__global__ __launch_bounds__(128) void attn(
    const unsigned short* __restrict__ qws, const unsigned short* __restrict__ kws,
    const unsigned short* __restrict__ vws, const float* __restrict__ rpb,
    const float* __restrict__ mask, unsigned short* __restrict__ aout) {
  const unsigned bh = blockIdx.x, b = bh >> 4, h = bh & 15;
  __shared__ __align__(16) unsigned int lq[64 * 20], lk[64 * 20], lv[64 * 20];
  __shared__ __align__(16) unsigned int lp[4 * 16 * 36];  // per-wave P strip 16x64 bf16

  const int t = threadIdx.x, wave = t >> 5, lane = t & 31, hf = lane >> 4, l16 = lane & 15;
  const uint4* qg = (const uint4*)qws;
  const uint4* kg = (const uint4*)kws;
  const uint4* vg = (const uint4*)vws;
  const unsigned tileq4 = bh * 256;   // 256 uint4 per (b,h) tensor tile
#pragma unroll
  for (int p = 0; p < 2; ++p) {
    int lin = p * 128 + t;
    int r = lin >> 2, q4 = lin & 3;
    unsigned goff = tileq4 + r * 4 + q4;
    *(uint4*)(lq + r * 20 + q4 * 4) = qg[goff];
    *(uint4*)(lk + r * 20 + q4 * 4) = kg[goff];
    *(uint4*)(lv + r * 20 + q4 * 4) = vg[goff];
  }
  __syncthreads();

  // S = q @ k^T  (K = hd = 32 -> one WMMA per 16x16 tile)
  v8f s[4];
#pragma unroll
  for (int j = 0; j < 4; ++j) s[j] = (v8f){};
  {
    v16bf qa = frag_a(lq, wave * 16, 20, 0);
#pragma unroll
    for (int j = 0; j < 4; ++j) {
      v16bf kb = frag_b(lk, j * 16, 20, 0);
      s[j] = wmma_bf16(qa, kb, s[j]);
    }
  }

  // + rpb + mask, row softmax (rows live in 16-lane halves for fixed acc elem)
  const float* rp = rpb + h * 4096;
  const float* mk = mask + (b & 63) * 4096;
#pragma unroll
  for (int e = 0; e < 8; ++e) {
    int i = wave * 16 + e + 8 * hf;
#pragma unroll
    for (int j = 0; j < 4; ++j) {
      int col = 16 * j + l16;
      s[j][e] += rp[i * 64 + col] + mk[i * 64 + col];
    }
    float mx = fmaxf(fmaxf(s[0][e], s[1][e]), fmaxf(s[2][e], s[3][e]));
#pragma unroll
    for (int xm = 1; xm <= 8; xm <<= 1) mx = fmaxf(mx, __shfl_xor(mx, xm, 32));
    float sum = 0.f;
#pragma unroll
    for (int j = 0; j < 4; ++j) {
      float ee = __expf(s[j][e] - mx);
      s[j][e] = ee;
      sum += ee;
    }
#pragma unroll
    for (int xm = 1; xm <= 8; xm <<= 1) sum += __shfl_xor(sum, xm, 32);
    float inv = 1.f / sum;
#pragma unroll
    for (int j = 0; j < 4; ++j) s[j][e] *= inv;
  }

  // Re-layout P through per-wave LDS strip (same-wave RAW; compiler inserts dscnt waits)
  unsigned short* lps = (unsigned short*)(lp + wave * 16 * 36);
#pragma unroll
  for (int j = 0; j < 4; ++j)
#pragma unroll
    for (int e = 0; e < 8; ++e)
      lps[(e + 8 * hf) * 72 + 16 * j + l16] = f2bf(s[j][e]);

  // O = P @ V  (K = 64 -> two 32-chunks; hd = 32 -> two 16-col tiles)
  v8f o[2] = {(v8f){}, (v8f){}};
#pragma unroll
  for (int kc = 0; kc < 2; ++kc) {
    v16bf pa = frag_a(lp + wave * 16 * 36, 0, 36, kc * 16);
#pragma unroll
    for (int ct = 0; ct < 2; ++ct) {
      v16bf vb = frag_b_strided((const unsigned short*)lv, 16 * ct + l16, 40, kc * 32);
      o[ct] = wmma_bf16(pa, vb, o[ct]);
    }
  }
#pragma unroll
  for (int ct = 0; ct < 2; ++ct)
#pragma unroll
    for (int e = 0; e < 8; ++e) {
      unsigned i = wave * 16 + e + 8 * hf;
      unsigned d = 16 * ct + l16;
      aout[(b * 64 + i) * 512 + h * 32 + d] = f2bf(o[ct][e]);
    }
}

// ---------------------------------------------------------------------------
// Kernel 4: out = attn_out(bf16) @ w_proj + b_proj  (M=131072, N=K=512)
// ---------------------------------------------------------------------------
__global__ __launch_bounds__(256) void proj_gemm(
    const unsigned short* __restrict__ aio, const float* __restrict__ w,
    const float* __restrict__ bias, float* __restrict__ out) {
  const unsigned N0 = (blockIdx.x % 4) * 128;
  const unsigned M0 = (blockIdx.x / 4) * 128;
  __shared__ __align__(16) unsigned int lds_a[128 * 20];
  __shared__ __align__(16) unsigned int lds_b[128 * 20];
  const int t = threadIdx.x;
  const int wave = t >> 5, wm = wave >> 1, wn = wave & 1;
  const int lane = t & 31, hf = lane >> 4, l16 = lane & 15;

  v8f acc[2][4];
#pragma unroll
  for (int mt = 0; mt < 2; ++mt)
#pragma unroll
    for (int nt = 0; nt < 4; ++nt) acc[mt][nt] = (v8f){};

#pragma unroll 1
  for (unsigned kc = 0; kc < 16; ++kc) {
    __syncthreads();
    // A tile: bf16 pass-through copy, uint4 (b128) loads/stores, 32-bit offsets
#pragma unroll
    for (int p = 0; p < 2; ++p) {
      int lin = p * 256 + t;
      int m = lin >> 2, q4 = lin & 3;
      unsigned goff = (M0 + m) * 64 + kc * 4 + q4;   // in uint4 units (row = 64 uint4)
      *(uint4*)(lds_a + m * 20 + q4 * 4) = ((const uint4*)aio)[goff];
    }
    load_b_fast(w, 512, N0, kc * 32, lds_b, t);
    __syncthreads();
    mma_128x128(lds_a, lds_b, wm, wn, acc);
  }
#pragma unroll
  for (int mt = 0; mt < 2; ++mt)
#pragma unroll
    for (int nt = 0; nt < 4; ++nt)
#pragma unroll
      for (int e = 0; e < 8; ++e) {
        unsigned m = M0 + wm * 32 + mt * 16 + e + 8 * hf;
        unsigned c = N0 + wn * 64 + nt * 16 + l16;
        out[m * 512 + c] = acc[mt][nt][e] + bias[c];
      }
}

// ---------------------------------------------------------------------------
extern "C" void kernel_launch(void* const* d_in, const int* in_sizes, int n_in,
                              void* d_out, int out_size, void* d_ws, size_t ws_size,
                              hipStream_t stream) {
  const float* x     = (const float*)d_in[0];
  const float* y     = (const float*)d_in[1];
  const float* mask  = (const float*)d_in[2];
  const float* w1    = (const float*)d_in[3];
  const float* b1    = (const float*)d_in[4];
  const float* w2    = (const float*)d_in[5];
  const float* b2    = (const float*)d_in[6];
  const float* table = (const float*)d_in[7];
  const int*   ridx  = (const int*)d_in[8];
  const float* wp    = (const float*)d_in[9];
  const float* bp    = (const float*)d_in[10];
  float* out = (float*)d_out;

  char* ws = (char*)d_ws;
  const size_t QKV = (size_t)2048 * 16 * 64 * 32 * 2;  // 128 MiB per tensor (bf16)
  unsigned short* qws  = (unsigned short*)(ws);
  unsigned short* kws  = (unsigned short*)(ws + QKV);
  unsigned short* vws  = (unsigned short*)(ws + 2 * QKV);
  unsigned short* aout = (unsigned short*)(ws + 3 * QKV);
  float*          rpb  = (float*)(ws + 4 * QKV);

  rpb_gather<<<16, 256, 0, stream>>>(table, ridx, rpb);
  qkv_gemm<<<1024 * 12, 256, 0, stream>>>(x, y, w1, b1, w2, b2, qws, kws, vws);
  attn<<<32768, 128, 0, stream>>>(qws, kws, vws, rpb, mask, aout);
  proj_gemm<<<1024 * 4, 256, 0, stream>>>(aout, wp, bp, out);
}
